// BoltzmannMachine_48163763257845
// MI455X (gfx1250) — matrix operand order repaired and checked by reading, not compiled, not run on hardware
//
#include <hip/hip_runtime.h>
#include <hip/hip_bf16.h>

// ---------------- problem constants ----------------
#define IN_DIM   1024
#define OUT_DIM  1024
#define HID_DIM  2048
#define L_DIM    4096              // IN + OUT + HID
#define B_DIM    1024              // batch
#define NCOMP    (L_DIM - IN_DIM)  // 3072 computed output columns
#define N_ITERS  32
#define EPS_NRM  1e-12f

// WMMA tiling
#define BK 32                      // K per WMMA (bf16)
#define BM 128                     // block tile M
#define BN 128                     // block tile N

typedef __attribute__((ext_vector_type(16))) __bf16 v16bf;
typedef __attribute__((ext_vector_type(8)))  float  v8f;

// ---------------------------------------------------------------------------
// CDNA5 async copy: global -> LDS, 16B per lane, tracked by ASYNCcnt.
// Low 32 bits of a flat shared-aperture address are the LDS byte offset
// (see ISA section 10.2).
// ---------------------------------------------------------------------------
__device__ __forceinline__ void async_b128(void* ldsDst, const void* gSrc)
{
    unsigned loff = (unsigned)(uintptr_t)ldsDst;
    asm volatile("global_load_async_to_lds_b128 %0, %1, off"
                 :: "v"(loff), "v"(gSrc)
                 : "memory");
}

// ---------------------------------------------------------------------------
// Prep 1: mwbf[j*L + k] = bf16( W[j,k] * A[k,j] )   (row j = B-operand column j)
// ---------------------------------------------------------------------------
__global__ __launch_bounds__(256)
void prep_mw_kernel(const float* __restrict__ W, const float* __restrict__ A,
                    __bf16* __restrict__ mwbf)
{
    size_t idx = (size_t)blockIdx.x * 256 + threadIdx.x;   // j*L + k
    int j = (int)(idx >> 12);
    int k = (int)(idx & (L_DIM - 1));
    float v = W[idx] * A[(size_t)k * L_DIM + j];
    mwbf[idx] = (__bf16)v;
}

// ---------------------------------------------------------------------------
// Prep 2: act f32 (=d_out) and bf16 mirror: cols [0,IN)=x, rest 0
// ---------------------------------------------------------------------------
__global__ __launch_bounds__(256)
void prep_act_kernel(const float* __restrict__ x, float* __restrict__ out,
                     __bf16* __restrict__ actbf)
{
    size_t idx = (size_t)blockIdx.x * 256 + threadIdx.x;   // row*L + col
    int col = (int)(idx & (L_DIM - 1));
    int row = (int)(idx >> 12);
    float v = (col < IN_DIM) ? x[(size_t)row * IN_DIM + col] : 0.0f;
    out[idx]   = v;
    actbf[idx] = (__bf16)v;
}

// ---------------------------------------------------------------------------
// GEMM: out[:, IN:L] = relu( actbf(BxL) @ mwbf(LxL, row j = col j of B) )
// 256 threads = 8 waves; block tile 128x128; wave tile 64x32 (4x2 WMMA tiles).
// Double-buffered LDS fed by global_load_async_to_lds_b128.
// ---------------------------------------------------------------------------
__global__ __launch_bounds__(256)
void gemm_relu_kernel(const __bf16* __restrict__ actbf,
                      const __bf16* __restrict__ mwbf,
                      float* __restrict__ out)
{
    __shared__ __bf16 lA[2][BM][BK];   // act tiles   [buf][m][k]
    __shared__ __bf16 lB[2][BN][BK];   // weight tiles[buf][n][k]

    const int tid   = threadIdx.x;
    const int lane  = tid & 31;
    const int wave  = tid >> 5;
    const int waveM = wave >> 2;              // 0..1 -> 64-row half
    const int waveN = wave & 3;               // 0..3 -> 32-col strip
    const int mBase = blockIdx.y * BM;
    const int nBase = IN_DIM + blockIdx.x * BN;

    // tile-load mapping: 512 x 16B chunks, thread t -> chunks t, t+256
    const int c0 = tid, c1 = tid + 256;
    const int r0 = c0 >> 2, o0 = (c0 & 3) * 8;
    const int r1 = c1 >> 2, o1 = (c1 & 3) * 8;

    const __bf16* gA0 = &actbf[(size_t)(mBase + r0) * L_DIM + o0];
    const __bf16* gA1 = &actbf[(size_t)(mBase + r1) * L_DIM + o1];
    const __bf16* gB0 = &mwbf [(size_t)(nBase + r0) * L_DIM + o0];
    const __bf16* gB1 = &mwbf [(size_t)(nBase + r1) * L_DIM + o1];

    v8f acc[4][2] = {};
    const int khalf = lane >> 4;    // 0/1: which K half this lane holds
    const int lrow  = lane & 15;

    // issue chunk 0 into buffer 0
    async_b128(&lA[0][r0][o0], gA0);
    async_b128(&lA[0][r1][o1], gA1);
    async_b128(&lB[0][r0][o0], gB0);
    async_b128(&lB[0][r1][o1], gB1);

    const int NCHUNK = L_DIM / BK;            // 128
    for (int kc = 0; kc < NCHUNK; ++kc) {
        const int cur = kc & 1, nxt = cur ^ 1;
        if (kc + 1 < NCHUNK) {
            // issue next chunk into the other buffer, then wait for the
            // current chunk's 4 loads (async loads complete in order).
            const int ko = (kc + 1) * BK;
            async_b128(&lA[nxt][r0][o0], gA0 + ko);
            async_b128(&lA[nxt][r1][o1], gA1 + ko);
            async_b128(&lB[nxt][r0][o0], gB0 + ko);
            async_b128(&lB[nxt][r1][o1], gB1 + ko);
            asm volatile("s_wait_asynccnt 0x4" ::: "memory");
        } else {
            asm volatile("s_wait_asynccnt 0x0" ::: "memory");
        }
        __syncthreads();   // current-chunk LDS data visible to all waves

        // ---- build fragments per ISA 7.12.2 layouts ----
        // A 16x32: lane m (0-15) K=0..7,16..23 ; lane m+16: K=8..15,24..31
        v16bf afr[4];
        #pragma unroll
        for (int t = 0; t < 4; ++t) {
            const __bf16* src = &lA[cur][waveM * 64 + t * 16 + lrow][0];
            ((uint4*)&afr[t])[0] = *(const uint4*)(src + khalf * 8);
            ((uint4*)&afr[t])[1] = *(const uint4*)(src + 16 + khalf * 8);
        }
        // B 32x16: lanes 0-15 col n, K=0..7,16..23 ; lanes 16-31 col n, K=8..15,24..31
        v16bf bfr[2];
        #pragma unroll
        for (int u = 0; u < 2; ++u) {
            const __bf16* src = &lB[cur][waveN * 32 + u * 16 + lrow][0];
            ((uint4*)&bfr[u])[0] = *(const uint4*)(src + khalf * 8);
            ((uint4*)&bfr[u])[1] = *(const uint4*)(src + 16 + khalf * 8);
        }

        // ---- 8 WMMAs per wave per K-chunk ----
        #pragma unroll
        for (int t = 0; t < 4; ++t)
            #pragma unroll
            for (int u = 0; u < 2; ++u)
                acc[t][u] = __builtin_amdgcn_wmma_f32_16x16x32_bf16(
                    false, afr[t], false, bfr[u],
                    (short)0, acc[t][u], false, false);

        __syncthreads();   // all waves done reading before buf is overwritten
    }

    // ---- epilogue: relu + store f32 (C/D layout: VGPR i -> M=i / i+8) ----
    const int mAdd = khalf * 8;
    #pragma unroll
    for (int t = 0; t < 4; ++t) {
        #pragma unroll
        for (int u = 0; u < 2; ++u) {
            const int gc = nBase + waveN * 32 + u * 16 + lrow;
            #pragma unroll
            for (int i = 0; i < 8; ++i) {
                float v = acc[t][u][i];
                v = v > 0.0f ? v : 0.0f;
                const int gr = mBase + waveM * 64 + t * 16 + i + mAdd;
                out[(size_t)gr * L_DIM + gc] = v;
            }
        }
    }
}

// ---------------------------------------------------------------------------
// Row norm of hidden section: nrm[row] = ||out[row, IN+OUT:]||_2
// ---------------------------------------------------------------------------
__global__ __launch_bounds__(256)
void rownorm_kernel(const float* __restrict__ out, float* __restrict__ nrm)
{
    const int row = blockIdx.x;
    const float* p = out + (size_t)row * L_DIM + IN_DIM + OUT_DIM;
    float s = 0.0f;
    for (int c = threadIdx.x; c < HID_DIM; c += 256) {
        float v = p[c];
        s += v * v;
    }
    #pragma unroll
    for (int off = 16; off > 0; off >>= 1)
        s += __shfl_down(s, off, 32);
    __shared__ float red[8];
    if ((threadIdx.x & 31) == 0) red[threadIdx.x >> 5] = s;
    __syncthreads();
    if (threadIdx.x == 0) {
        float t = 0.0f;
        #pragma unroll
        for (int i = 0; i < 8; ++i) t += red[i];
        nrm[row] = sqrtf(t);
    }
}

// ---------------------------------------------------------------------------
// Scale hidden cols by 1/max(nrm,eps); refresh bf16 mirror of cols [IN, L)
// ---------------------------------------------------------------------------
__global__ __launch_bounds__(256)
void scale_cvt_kernel(float* __restrict__ out, __bf16* __restrict__ actbf,
                      const float* __restrict__ nrm)
{
    int idx = blockIdx.x * 256 + threadIdx.x;   // over B_DIM * NCOMP
    int row = idx / NCOMP;
    int c   = idx - row * NCOMP;
    int gc  = IN_DIM + c;
    size_t off = (size_t)row * L_DIM + gc;
    float v = out[off];
    if (c >= OUT_DIM) {                          // hidden section
        v = v / fmaxf(nrm[row], EPS_NRM);
        out[off] = v;
    }
    actbf[off] = (__bf16)v;
}

// ---------------------------------------------------------------------------
extern "C" void kernel_launch(void* const* d_in, const int* in_sizes, int n_in,
                              void* d_out, int out_size, void* d_ws, size_t ws_size,
                              hipStream_t stream)
{
    (void)in_sizes; (void)n_in; (void)out_size; (void)ws_size;
    const float* x = (const float*)d_in[0];
    // d_in[1] = y (unused by the reference computation)
    const float* W = (const float*)d_in[2];
    const float* A = (const float*)d_in[3];
    // d_in[4] = n (device scalar; reference fixes n = 32)

    float* out = (float*)d_out;
    char*  ws  = (char*)d_ws;
    __bf16* mwbf  = (__bf16*)ws;                                       // 32 MB
    __bf16* actbf = (__bf16*)(ws + (size_t)L_DIM * L_DIM * 2);         //  8 MB
    float*  nrm   = (float*)(ws + (size_t)L_DIM * L_DIM * 2
                                + (size_t)B_DIM * L_DIM * 2);          //  4 KB

    // one-time prep (inside graph: re-runs deterministically each replay)
    prep_mw_kernel <<< (L_DIM * (size_t)L_DIM) / 256, 256, 0, stream >>> (W, A, mwbf);
    prep_act_kernel<<< (B_DIM * (size_t)L_DIM) / 256, 256, 0, stream >>> (x, out, actbf);

    dim3 gemm_grid(NCOMP / BN, B_DIM / BM);   // 24 x 8 workgroups
    for (int it = 0; it < N_ITERS; ++it) {
        gemm_relu_kernel<<< gemm_grid, 256, 0, stream >>>(actbf, mwbf, out);
        rownorm_kernel  <<< B_DIM, 256, 0, stream >>>(out, nrm);
        scale_cvt_kernel<<< (B_DIM * NCOMP) / 256, 256, 0, stream >>>(out, actbf, nrm);
    }
}
